// MHA_47467978556069
// MI455X (gfx1250) — matrix operand order repaired and checked by reading, not compiled, not run on hardware
//
#include <hip/hip_runtime.h>
#include <hip/hip_bf16.h>

// MHA forward, MI455X / gfx1250, wave32, bf16 WMMA with f32 accumulate.
// Sizes fixed by the reference.
#define B_   2
#define S_   2048
#define DIN  256
#define E_   256
#define H_   8
#define HD_  32

typedef __attribute__((ext_vector_type(16))) __bf16 v16bf;
typedef __attribute__((ext_vector_type(8)))  __bf16 v8bf;
typedef __attribute__((ext_vector_type(8)))  float  v8f;
typedef __attribute__((ext_vector_type(4)))  int    v4i;

// gfx1250 async global->LDS path (ASYNCcnt-tracked), with guarded fallback.
#if defined(__gfx1250__) && \
    __has_builtin(__builtin_amdgcn_global_load_async_to_lds_b128) && \
    __has_builtin(__builtin_amdgcn_s_wait_asynccnt)
#define USE_ASYNC_LDS 1
#else
#define USE_ASYNC_LDS 0
#endif

static __device__ __forceinline__ v8f wmma_bf16(v16bf a, v16bf b, v8f c) {
  // D = A(16x32 bf16) * B(32x16 bf16) + C(16x16 f32)
  return __builtin_amdgcn_wmma_f32_16x16x32_bf16(
      /*neg_a=*/false, a, /*neg_b=*/false, b,
      /*c_mod=*/(short)0, c, /*reuse_a=*/false, /*reuse_b=*/false);
}

// Load a 16-element bf16 fragment as two 16B-aligned 8-element chunks.
// off2 = 8 for both A- and B-fragments here: 16 contiguous K per lane half.
static __device__ __forceinline__ v16bf load_frag(const __bf16* p, int off2) {
  v8bf lo = *(const v8bf*)(p);
  v8bf hi = *(const v8bf*)(p + off2);
  return __builtin_shufflevector(lo, hi, 0,1,2,3,4,5,6,7,8,9,10,11,12,13,14,15);
}

// ---------------------------------------------------------------------------
// Kernel 1: fp32 -> bf16 conversions and weight transposes.
//   xb    [B*S, DIN]      = bf16(x)
//   Wqkvt [3E, DIN]       = bf16(Wqkv^T)   (row n holds K contiguous)
//   Wot   [E, E]          = bf16(Wo^T)
// ---------------------------------------------------------------------------
__global__ __launch_bounds__(256) void prep_kernel(
    const float* __restrict__ x, const float* __restrict__ Wqkv,
    const float* __restrict__ Wo,
    __bf16* __restrict__ xb, __bf16* __restrict__ Wqkvt, __bf16* __restrict__ Wot) {
  int i = blockIdx.x * 256 + threadIdx.x;
  const int NX  = B_ * S_ * DIN;   // 1048576
  const int NW1 = 3 * E_ * DIN;    // 196608
  const int NW2 = E_ * E_;         // 65536
  if (i < NX) { xb[i] = (__bf16)x[i]; return; }
  i -= NX;
  if (i < NW1) { int n = i / DIN, k = i % DIN; Wqkvt[i] = (__bf16)Wqkv[(size_t)k * (3*E_) + n]; return; }
  i -= NW1;
  if (i < NW2) { int n = i / E_, k = i % E_; Wot[i] = (__bf16)Wo[(size_t)k * E_ + n]; return; }
}

// ---------------------------------------------------------------------------
// Kernel 2: QKV projection GEMM [4096,256] x [256,768], WMMA bf16.
// One 16x16 output tile per wave. Scatter into:
//   Qh [bh][s][hd]  (pre-scaled by 1/sqrt(HD))
//   Kh [bh][s][hd]
//   Vt [bh][hd][s]  (transposed for contiguous P*V B-fragments)
// ---------------------------------------------------------------------------
__global__ __launch_bounds__(256) void qkv_kernel(
    const __bf16* __restrict__ xb, const __bf16* __restrict__ Wqkvt,
    const float* __restrict__ bqkv,
    __bf16* __restrict__ Qh, __bf16* __restrict__ Kh, __bf16* __restrict__ Vt) {
  const int lane = threadIdx.x & 31;
  const int wave = threadIdx.x >> 5;
  const int NT = (3 * E_) / 16;                 // 48 column tiles
  int tile = blockIdx.x * 8 + wave;
  int mt = tile / NT, nt = tile % NT;

  int arow = mt * 16 + (lane & 15);
  int n    = nt * 16 + (lane & 15);
  const __bf16* ap = xb    + (size_t)arow * DIN + ((lane >> 4) << 4);
  const __bf16* bp = Wqkvt + (size_t)n    * DIN + ((lane >> 4) << 4);

  v8f acc = {};
  #pragma unroll
  for (int kk = 0; kk < DIN; kk += 32) {
    v16bf a = load_frag(ap + kk, 8);
    v16bf b = load_frag(bp + kk, 8);
    acc = wmma_bf16(a, b, acc);
  }

  float bias = bqkv[n];
  int h = n / 96, j = n % 96, sec = j >> 5, hd = j & 31;
  #pragma unroll
  for (int r = 0; r < 8; r++) {
    int m  = mt * 16 + r + ((lane >> 4) << 3);  // C layout: M = r + 8*(lane/16)
    int b_ = m >> 11;                            // / S_
    int s  = m & (S_ - 1);
    size_t bh = (size_t)b_ * H_ + h;
    float v = acc[r] + bias;
    if (sec == 0)      Qh[(bh * S_ + s) * HD_ + hd] = (__bf16)(v * 0.17677669529663689f);
    else if (sec == 1) Kh[(bh * S_ + s) * HD_ + hd] = (__bf16)v;
    else               Vt[(bh * HD_ + hd) * S_ + s] = (__bf16)v;
  }
}

// ---------------------------------------------------------------------------
// Kernel 3: fused flash attention with weighted softmax.
// Wave = 16 queries; loop keys in 32-step tiles:
//   weights tile -> LDS via ASYNC global->LDS (overlaps QK^T WMMAs)
//   S = Q*K^T (2 WMMA) -> online max/exp -> *weights[b][k][q] -> P in LDS (bf16)
//   O += P*V (2 WMMA), online rescale; final O/l -> ctx bf16 [b][s][h*HD+hd]
// ---------------------------------------------------------------------------
__global__ __launch_bounds__(256) void attn_kernel(
    const __bf16* __restrict__ Qh, const __bf16* __restrict__ Kh,
    const __bf16* __restrict__ Vt, const float* __restrict__ weights,
    __bf16* __restrict__ ctx) {
  // Per-wave private LDS. wtile rows padded to 20 floats -> conflict-free
  // column reads (20*l mod 64 distinct for l=0..15) and 16B-aligned rows (80B).
  __shared__ __align__(16) float  wtile[8][32][20];   // 20480 B
  __shared__ __align__(16) __bf16 ptile[8][16][32];   //  8192 B

  const int lane = threadIdx.x & 31;
  const int wave = threadIdx.x >> 5;
  const int blocksPerBH = S_ / 128;                   // 16
  const int bh = blockIdx.x / blocksPerBH;
  const int q0 = (blockIdx.x % blocksPerBH) * 128 + wave * 16;
  const int b  = bh / H_;
  const int h  = bh % H_;

  // Q A-fragment: lane row = lane%16, K contiguous (lane/16)*16 .. +15.
  const __bf16* qp = Qh + ((size_t)bh * S_ + q0 + (lane & 15)) * HD_ + ((lane >> 4) << 4);
  v16bf qa = load_frag(qp, 8);

  v8f o0 = {}, o1 = {};
  float mrun[8], lsum[8];
  #pragma unroll
  for (int r = 0; r < 8; r++) { mrun[r] = -3.0e38f; lsum[r] = 0.0f; }

  const float*  wbase = weights + (size_t)b * S_ * S_ + q0;
  const __bf16* kbase = Kh + (size_t)bh * S_ * HD_;
  const __bf16* vbase = Vt + (size_t)bh * HD_ * S_;

  for (int kt = 0; kt < S_; kt += 32) {
    // --- stage weights tile [32 keys][16 queries] into this wave's LDS slab ---
    {
      const float* wsrc = wbase + (size_t)(kt + lane) * S_;   // 64B row per lane
      float*       wdst = &wtile[wave][lane][0];
#if USE_ASYNC_LDS
      // Direct global->LDS async copy, tracked by ASYNCcnt; overlaps the
      // K-fragment loads and QK^T WMMAs below.
      #pragma unroll
      for (int c = 0; c < 4; c++) {
        __builtin_amdgcn_global_load_async_to_lds_b128(
            (v4i*)(wsrc + c * 4), (v4i*)(wdst + c * 4), 0, 0);
      }
#else
      float4 w0 = ((const float4*)wsrc)[0];
      float4 w1 = ((const float4*)wsrc)[1];
      float4 w2 = ((const float4*)wsrc)[2];
      float4 w3 = ((const float4*)wsrc)[3];
      ((float4*)wdst)[0] = w0; ((float4*)wdst)[1] = w1;
      ((float4*)wdst)[2] = w2; ((float4*)wdst)[3] = w3;
#endif
    }

    // --- S = Q * K^T : B-fragment lane = key column, dims contiguous ---
    const __bf16* kp = kbase + (size_t)(kt + (lane & 15)) * HD_ + ((lane >> 4) << 4);
    v16bf kb0 = load_frag(kp, 8);
    v16bf kb1 = load_frag(kp + 16 * HD_, 8);
    v8f s0 = {}, s1 = {};
    s0 = wmma_bf16(qa, kb0, s0);
    s1 = wmma_bf16(qa, kb1, s1);
    if (kt + 32 < S_) __builtin_prefetch(kp + 32 * HD_, 0, 1);

    // --- row max across the 16-lane half (xor masks <= 8 stay in-half) ---
    float mt8[8];
    #pragma unroll
    for (int r = 0; r < 8; r++) mt8[r] = fmaxf(s0[r], s1[r]);
    #pragma unroll
    for (int off = 8; off >= 1; off >>= 1) {
      #pragma unroll
      for (int r = 0; r < 8; r++) mt8[r] = fmaxf(mt8[r], __shfl_xor(mt8[r], off));
    }

    // Weights tile must be resident in LDS before the exp/multiply stage.
#if USE_ASYNC_LDS
    __builtin_amdgcn_s_wait_asynccnt(0);
#endif
    asm volatile("" ::: "memory");  // order async/LDS writes vs. ds reads

    float alpha[8], rs[8];
    #pragma unroll
    for (int r = 0; r < 8; r++) {
      float mn = fmaxf(mrun[r], mt8[r]);
      alpha[r] = __expf(mrun[r] - mn);
      mrun[r]  = mn;
      int M = r + ((lane >> 4) << 3);
      float wv0 = wtile[wave][lane & 15][M];          // weights[b][kt+N][q0+M]
      float wv1 = wtile[wave][16 + (lane & 15)][M];
      float p0 = __expf(s0[r] - mn) * wv0;
      float p1 = __expf(s1[r] - mn) * wv1;
      ptile[wave][M][lane & 15]        = (__bf16)p0;  // P tile 16x32 row-major
      ptile[wave][M][16 + (lane & 15)] = (__bf16)p1;
      rs[r] = p0 + p1;
    }
    #pragma unroll
    for (int off = 8; off >= 1; off >>= 1) {
      #pragma unroll
      for (int r = 0; r < 8; r++) rs[r] += __shfl_xor(rs[r], off);
    }
    #pragma unroll
    for (int r = 0; r < 8; r++) {
      lsum[r] = lsum[r] * alpha[r] + rs[r];
      o0[r] *= alpha[r];
      o1[r] *= alpha[r];
    }

    asm volatile("" ::: "memory");

    // --- O += P * V ---
    const __bf16* pp = &ptile[wave][lane & 15][(lane >> 4) << 4];
    v16bf pa = load_frag(pp, 8);
    const __bf16* vp = vbase + (size_t)(lane & 15) * S_ + kt + ((lane >> 4) << 4);
    v16bf vb0 = load_frag(vp, 8);            // dims 0..15
    v16bf vb1 = load_frag(vp + 16 * S_, 8);  // dims 16..31
    o0 = wmma_bf16(pa, vb0, o0);
    o1 = wmma_bf16(pa, vb1, o1);
    if (kt + 32 < S_) __builtin_prefetch(vp + 32, 0, 1);

    asm volatile("" ::: "memory");
  }

  #pragma unroll
  for (int r = 0; r < 8; r++) {
    float inv = 1.0f / lsum[r];
    int q = q0 + r + ((lane >> 4) << 3);
    size_t base = ((size_t)b * S_ + q) * E_ + h * HD_;
    ctx[base + (lane & 15)]      = (__bf16)(o0[r] * inv);
    ctx[base + 16 + (lane & 15)] = (__bf16)(o1[r] * inv);
  }
}

// ---------------------------------------------------------------------------
// Kernel 4: out = ctx @ Wo + bo, WMMA bf16 GEMM, f32 output.
// ---------------------------------------------------------------------------
__global__ __launch_bounds__(256) void out_kernel(
    const __bf16* __restrict__ ctx, const __bf16* __restrict__ Wot,
    const float* __restrict__ bo, float* __restrict__ out) {
  const int lane = threadIdx.x & 31;
  const int wave = threadIdx.x >> 5;
  const int NT = E_ / 16;                      // 16
  int tile = blockIdx.x * 8 + wave;
  int mt = tile / NT, nt = tile % NT;

  const __bf16* ap = ctx + (size_t)(mt * 16 + (lane & 15)) * E_ + ((lane >> 4) << 4);
  const __bf16* bp = Wot + (size_t)(nt * 16 + (lane & 15)) * E_ + ((lane >> 4) << 4);

  v8f acc = {};
  #pragma unroll
  for (int kk = 0; kk < E_; kk += 32) {
    acc = wmma_bf16(load_frag(ap + kk, 8), load_frag(bp + kk, 8), acc);
  }
  float bias = bo[nt * 16 + (lane & 15)];
  #pragma unroll
  for (int r = 0; r < 8; r++) {
    int m = mt * 16 + r + ((lane >> 4) << 3);
    out[(size_t)m * E_ + nt * 16 + (lane & 15)] = acc[r] + bias;   // 64B coalesced
  }
}

// ---------------------------------------------------------------------------
extern "C" void kernel_launch(void* const* d_in, const int* in_sizes, int n_in,
                              void* d_out, int out_size, void* d_ws, size_t ws_size,
                              hipStream_t stream) {
  const float* x       = (const float*)d_in[0];
  const float* weights = (const float*)d_in[1];
  const float* Wqkv    = (const float*)d_in[2];
  const float* bqkv    = (const float*)d_in[3];
  const float* Wo      = (const float*)d_in[4];
  const float* bo      = (const float*)d_in[5];
  float* out = (float*)d_out;

  // Workspace layout (bytes), ~11 MB total.
  char* ws = (char*)d_ws;
  __bf16* xb    = (__bf16*)(ws + 0);         // 4096*256*2   = 2 MB
  __bf16* Wqkvt = (__bf16*)(ws + 2097152);   // 768*256*2    = 384 KB
  __bf16* Wot   = (__bf16*)(ws + 2490368);   // 256*256*2    = 128 KB
  __bf16* Qh    = (__bf16*)(ws + 2621440);   // 16*2048*32*2 = 2 MB
  __bf16* Kh    = (__bf16*)(ws + 4718592);   // 2 MB
  __bf16* Vt    = (__bf16*)(ws + 6815744);   // 2 MB
  __bf16* ctx   = (__bf16*)(ws + 8912896);   // 2 MB

  const int prepN = B_ * S_ * DIN + 3 * E_ * DIN + E_ * E_;   // 1,310,720
  prep_kernel<<<(prepN + 255) / 256, 256, 0, stream>>>(x, Wqkv, Wo, xb, Wqkvt, Wot);

  // 256 row-tiles * 48 col-tiles, 8 wave-tiles per block
  qkv_kernel<<<(256 * 48) / 8, 256, 0, stream>>>(xb, Wqkvt, bqkv, Qh, Kh, Vt);

  // B*H*(S/128) = 256 blocks, 8 waves each = 128 queries/block
  attn_kernel<<<B_ * H_ * (S_ / 128), 256, 0, stream>>>(Qh, Kh, Vt, weights, ctx);

  // 256 row-tiles * 16 col-tiles / 8
  out_kernel<<<(256 * 16) / 8, 256, 0, stream>>>(ctx, Wot, bo, out);
}
